// ResnetBlock_BigGAN_1597727834248
// MI455X (gfx1250) — matrix operand order repaired and checked by reading, not compiled, not run on hardware
//
#include <hip/hip_runtime.h>
#include <hip/hip_bf16.h>
#include <math.h>

namespace {

constexpr int    kB    = 4;
constexpr int    kN    = 49152;
constexpr int    kC    = 128;
constexpr int    kT    = 512;
constexpr int    kNNZ  = kN * 9;
constexpr int    kRows = kB * kN;            // 196608
constexpr size_t kE    = (size_t)kRows * kC; // 25165824 elements per [B,N,C] tensor
constexpr float  kEps  = 1e-5f;
constexpr int    kWp   = 4 * 2 * kC * 8;     // 8192 packed dwords per 128x128 weight matrix

typedef __attribute__((ext_vector_type(16))) __bf16       v16bf;
typedef __attribute__((ext_vector_type(8)))  float        v8f;
typedef __attribute__((ext_vector_type(8)))  unsigned int v8u;

__device__ __forceinline__ float silu_f(float x) { return x / (1.0f + expf(-x)); }
__device__ __forceinline__ float mish_f(float x) {
  float sp = (x > 20.0f) ? x : log1pf(expf(x));
  return x * tanhf(sp);
}
__device__ __forceinline__ unsigned int pack_bf16(float lo, float hi) {
  unsigned int l = __builtin_bit_cast(unsigned short, (__bf16)lo);
  unsigned int h = __builtin_bit_cast(unsigned short, (__bf16)hi);
  return l | (h << 16);
}

// ---- tiny time-MLP: t = silu(te) @ W + b for both MLPs at once ----
__global__ __launch_bounds__(128) void time_mlp_kernel(
    const float* __restrict__ te, const float* __restrict__ w1, const float* __restrict__ b1,
    const float* __restrict__ w2, const float* __restrict__ b2,
    float* __restrict__ t1, float* __restrict__ t2) {
  int b = blockIdx.x, c = threadIdx.x;
  float s1 = 0.f, s2 = 0.f;
  for (int j = 0; j < kT; ++j) {
    float v = silu_f(te[b * kT + j]);
    s1 += v * w1[j * kC + c];
    s2 += v * w2[j * kC + c];
  }
  t1[b * kC + c] = s1 + b1[c];
  t2[b * kC + c] = s2 + b2[c];
}

// ---- one-time weight pre-swizzle into WMMA B-fragment dword order ----
// Wp[(ks,half,ncol)*8 + p] = pack(bf16 W[k][ncol], bf16 W[k+1][ncol]), k = ks*32+half*8+2p+(p>=4?8:0)
__global__ __launch_bounds__(128) void pack_w_kernel(
    const float* __restrict__ W, unsigned int* __restrict__ Wp) {
  const float* Wk = W + (size_t)blockIdx.x * kC * kC;
  unsigned int* Wpk = Wp + (size_t)blockIdx.x * kWp;
  for (int d = threadIdx.x; d < kWp; d += 128) {
    int ncol = d & (kC - 1);
    int p    = (d >> 7) & 7;
    int half = (d >> 10) & 1;
    int ks   = d >> 11;
    int k    = ks * 32 + half * 8 + 2 * p + (p >= 4 ? 8 : 0);
    Wpk[(((ks * 2 + half) * kC) + ncol) * 8 + p] =
        pack_bf16(Wk[(size_t)k * kC + ncol], Wk[(size_t)(k + 1) * kC + ncol]);
  }
}

// ---- batchnorm stats: per-channel sum / sumsq of (src + t[b]) ----
// 256 rows per block; 256 divides N, so the whole block sits in one batch.
constexpr int kStatRows = 256;
__global__ __launch_bounds__(128) void bn_stats_kernel(
    const float* __restrict__ src, const float* __restrict__ tvec, float* __restrict__ stats) {
  int c = threadIdx.x;
  size_t r0 = (size_t)blockIdx.x * kStatRows;
  int b = (int)(r0 / kN);
  float tb = tvec[b * kC + c];
  float s = 0.f, s2 = 0.f;
  for (int i = 0; i < kStatRows; ++i) {
    float v = src[(r0 + (size_t)i) * kC + c] + tb;
    s += v; s2 += v * v;
  }
  atomicAdd(&stats[c], s);
  atomicAdd(&stats[kC + c], s2);
}

// ---- fused (add t) -> batchnorm -> mish, float4 per thread ----
__global__ __launch_bounds__(256) void bn_apply_kernel(
    const float* __restrict__ src, const float* __restrict__ tvec,
    const float* __restrict__ stats, const float* __restrict__ gamma,
    const float* __restrict__ beta, float* __restrict__ dst) {
  size_t idx = ((size_t)blockIdx.x * 256 + threadIdx.x) * 4;
  int c = (int)(idx & (kC - 1));
  int b = (int)(((size_t)blockIdx.x * 8) / kN);  // 8 rows per block, batch-aligned
  const float invM = 1.0f / (float)kRows;
  const float4 tv = *(const float4*)(tvec + b * kC + c);
  float4 v = *(const float4*)(src + idx);
  float4 o;
  {
    float m = stats[c + 0] * invM, va = stats[kC + c + 0] * invM - m * m;
    float w = (v.x + tv.x - m) * rsqrtf(va + kEps) * gamma[c + 0] + beta[c + 0];
    o.x = mish_f(w);
  }
  {
    float m = stats[c + 1] * invM, va = stats[kC + c + 1] * invM - m * m;
    float w = (v.y + tv.y - m) * rsqrtf(va + kEps) * gamma[c + 1] + beta[c + 1];
    o.y = mish_f(w);
  }
  {
    float m = stats[c + 2] * invM, va = stats[kC + c + 2] * invM - m * m;
    float w = (v.z + tv.z - m) * rsqrtf(va + kEps) * gamma[c + 2] + beta[c + 2];
    o.z = mish_f(w);
  }
  {
    float m = stats[c + 3] * invM, va = stats[kC + c + 3] * invM - m * m;
    float w = (v.w + tv.w - m) * rsqrtf(va + kEps) * gamma[c + 3] + beta[c + 3];
    o.w = mish_f(w);
  }
  *(float4*)(dst + idx) = o;
}

// ---- initialize an accumulator with broadcast bias (optionally two biases) ----
__global__ __launch_bounds__(256) void bias_init_kernel(
    float* __restrict__ out, const float* __restrict__ bias0, const float* __restrict__ bias1) {
  size_t idx = ((size_t)blockIdx.x * 256 + threadIdx.x) * 4;
  int c = (int)(idx & (kC - 1));
  float4 v = *(const float4*)(bias0 + c);
  if (bias1) {
    float4 u = *(const float4*)(bias1 + c);
    v.x += u.x; v.y += u.y; v.z += u.z; v.w += u.w;
  }
  *(float4*)(out + idx) = v;
}

// ---- sparse L @ y : out[:, rows[e]] += vals[e] * y[:, cols[e]] ----
// one wave32 per edge; each lane handles 4 channels (float4 gather, 4 f32 atomics);
// all 4 batches reuse the edge metadata. y and out are L2-resident (100MB << 192MB L2).
__global__ __launch_bounds__(256) void lmul_kernel(
    const float* __restrict__ y, const float* __restrict__ vals,
    const int* __restrict__ rows, const int* __restrict__ cols, float* __restrict__ out) {
  int e = (int)(((size_t)blockIdx.x * blockDim.x + threadIdx.x) >> 5);
  if (e >= kNNZ) return;
  int lane = threadIdx.x & 31;
  int r = rows[e], cidx = cols[e];
  float v = vals[e];
  int c4 = lane * 4;
#pragma unroll
  for (int b = 0; b < kB; ++b) {
    const float4 d = *(const float4*)(y + ((size_t)b * kN + cidx) * kC + c4);
    float* op = out + ((size_t)b * kN + r) * kC + c4;
    atomicAdd(op + 0, v * d.x);
    atomicAdd(op + 1, v * d.y);
    atomicAdd(op + 2, v * d.z);
    atomicAdd(op + 3, v * d.w);
  }
}

// ---- fused Chebyshev step + WMMA GEMM ----
//   recur==0: X = Xin
//   recur==1: X = 2*T - P, also written to Xout (next recurrence operand)
//   Out (M x 128) += X (M x 128) @ Wp (pre-swizzled 128x128), bf16 WMMA, f32 accumulate.
//
// Operands are staged in LDS pre-packed as bf16 pairs in exact WMMA fragment
// order -> every fragment is two aligned ds_load_b128. The accumulators are
// transposed through LDS so the global accumulate is coalesced float4 RMW.
constexpr int kTM     = 64;   // rows per workgroup: 4 waves x 16-row strips
constexpr int kXPitch = 68;   // dword pitch per row: 64 pairs + 4 pad (16B aligned, 4-bank rotate)
constexpr int kOPitch = 132;  // f32 pitch for the output transpose (half-wave bank split)
constexpr int kSMemB  = (kTM * kXPitch + kWp) * 4;  // 50176 B (>= kTM*kOPitch*4 = 33792)
__global__ __launch_bounds__(128) void cheb_gemm_kernel(
    const float* __restrict__ Xin, const float* __restrict__ T,
    const float* __restrict__ P, float* __restrict__ Xout,
    const unsigned int* __restrict__ Wp, float* __restrict__ Out, int recur) {
  __shared__ __align__(16) unsigned char SMem[kSMemB];
  unsigned int* Xp = (unsigned int*)SMem;            // packed X tile (K pairs)
  unsigned int* Bp = Xp + kTM * kXPitch;             // B fragments: [ks][half][ncol][8]
  int tid = threadIdx.x;
  size_t row0 = (size_t)blockIdx.x * kTM;

  // Stage pre-swizzled weights: pure b128 copy (L2-resident, 32KB).
  for (int d = tid * 4; d < kWp; d += 128 * 4)
    *(uint4*)(Bp + d) = *(const uint4*)(Wp + d);
  // Stage X tile as packed bf16 K-pairs, computing the Chebyshev recurrence on
  // the fly (and writing the f32 result back for the next L-multiply).
  for (int i = tid; i < kTM * 64; i += 128) {
    int r = i >> 6, cpair = i & 63;
    size_t g = (row0 + (size_t)r) * kC + 2 * (size_t)cpair;
    float2 v;
    if (recur) {
      float2 tv = *(const float2*)(T + g);
      float2 pv = *(const float2*)(P + g);
      v.x = 2.0f * tv.x - pv.x;
      v.y = 2.0f * tv.y - pv.y;
      *(float2*)(Xout + g) = v;
    } else {
      v = *(const float2*)(Xin + g);
    }
    Xp[r * kXPitch + cpair] = pack_bf16(v.x, v.y);
  }
  __syncthreads();

  int wave = tid >> 5;
  int lane = tid & 31;
  int half = lane >> 4;                      // ISA: lanes>=16 hold the upper K interleave
  int mrow = wave * 16 + (lane & 15);
  const unsigned int* xrow = Xp + mrow * kXPitch;

  v8f acc[8] = {};
#pragma unroll
  for (int ks = 0; ks < 4; ++ks) {           // K = 128 in 4 steps of 32
    // A fragment: pairs {a0..a0+3} and {a0+8..a0+11}  ->  2x ds_load_b128
    int a0 = ks * 16 + half * 4;
    uint4 alo = *(const uint4*)(xrow + a0);
    uint4 ahi = *(const uint4*)(xrow + a0 + 8);
    v8u au;
    au[0] = alo.x; au[1] = alo.y; au[2] = alo.z; au[3] = alo.w;
    au[4] = ahi.x; au[5] = ahi.y; au[6] = ahi.z; au[7] = ahi.w;
    v16bf a = __builtin_bit_cast(v16bf, au);
#pragma unroll
    for (int nt = 0; nt < 8; ++nt) {         // 8 output column tiles of 16
      int ncol = nt * 16 + (lane & 15);
      const unsigned int* bp = Bp + (((ks * 2 + half) * kC) + ncol) * 8;
      uint4 blo = *(const uint4*)(bp);
      uint4 bhi = *(const uint4*)(bp + 4);
      v8u bu;
      bu[0] = blo.x; bu[1] = blo.y; bu[2] = blo.z; bu[3] = blo.w;
      bu[4] = bhi.x; bu[5] = bhi.y; bu[6] = bhi.z; bu[7] = bhi.w;
      v16bf bf = __builtin_bit_cast(v16bf, bu);
      acc[nt] = __builtin_amdgcn_wmma_f32_16x16x32_bf16(
          false, a, false, bf, (short)0, acc[nt], false, false);
    }
  }

  // Transpose the D fragments through LDS (reusing the staging buffer), then
  // accumulate into Out with fully coalesced float4 RMW.
  __syncthreads();                           // everyone is done reading Xp/Bp
  float* OutS = (float*)SMem;
  int roff = half * 8;
  int colb = lane & 15;
#pragma unroll
  for (int nt = 0; nt < 8; ++nt)
#pragma unroll
    for (int r = 0; r < 8; ++r)
      OutS[(wave * 16 + r + roff) * kOPitch + nt * 16 + colb] = acc[nt][r];
  __syncthreads();
  for (int i = tid; i < kTM * 32; i += 128) {  // 32 float4 per row
    int r2 = i >> 5, c4 = (i & 31) * 4;
    size_t g = (row0 + (size_t)r2) * kC + (size_t)c4;
    float4 o = *(const float4*)(Out + g);
    const float4 s = *(const float4*)(OutS + r2 * kOPitch + c4);
    o.x += s.x; o.y += s.y; o.z += s.z; o.w += s.w;
    *(float4*)(Out + g) = o;
  }
}

}  // namespace

extern "C" void kernel_launch(void* const* d_in, const int* in_sizes, int n_in,
                              void* d_out, int out_size, void* d_ws, size_t ws_size,
                              hipStream_t stream) {
  (void)in_sizes; (void)n_in; (void)out_size; (void)ws_size;
  const float* x        = (const float*)d_in[0];
  const float* time_emb = (const float*)d_in[1];
  const float* w_mlp1   = (const float*)d_in[2];
  const float* b_mlp1   = (const float*)d_in[3];
  const float* w_mlp2   = (const float*)d_in[4];
  const float* b_mlp2   = (const float*)d_in[5];
  const float* gamma1   = (const float*)d_in[6];
  const float* beta1    = (const float*)d_in[7];
  const float* gamma2   = (const float*)d_in[8];
  const float* beta2    = (const float*)d_in[9];
  const float* w_conv1  = (const float*)d_in[10];
  const float* b_conv1  = (const float*)d_in[11];
  const float* w_conv2  = (const float*)d_in[12];
  const float* b_conv2  = (const float*)d_in[13];
  const float* w_res    = (const float*)d_in[14];
  const float* b_res    = (const float*)d_in[15];
  const float* lap_vals = (const float*)d_in[16];
  const int*   lap_rows = (const int*)d_in[17];
  const int*   lap_cols = (const int*)d_in[18];
  float* out = (float*)d_out;

  float* ws = (float*)d_ws;
  float* A  = ws;             // rotating Chebyshev buffer / normalized activations
  float* Bb = ws + kE;        // lmul scratch
  float* Cc = ws + 2 * kE;    // rotating Chebyshev buffer
  float* O1 = ws + 3 * kE;    // conv1 accumulator
  float* t1 = ws + 4 * kE;
  float* t2 = t1 + kB * kC;
  float* stats = t2 + kB * kC;
  unsigned int* wp1 = (unsigned int*)(stats + 2 * kC);  // 15 pre-swizzled weight mats
  unsigned int* wp2 = wp1 + 5 * kWp;
  unsigned int* wpr = wp2 + 5 * kWp;

  const int gridE4  = (int)(kE / 1024);
  const int gridBN  = kRows / kStatRows;
  const int gridGem = kRows / kTM;
  const int gridLm  = (int)(((size_t)kNNZ * 32 + 255) / 256);

  auto lmul = [&](const float* y, float* dst) {
    hipMemsetAsync(dst, 0, kE * sizeof(float), stream);
    lmul_kernel<<<gridLm, 256, 0, stream>>>(y, lap_vals, lap_rows, lap_cols, dst);
  };
  auto gemm_direct = [&](const float* X, const unsigned int* Wp, float* O) {
    cheb_gemm_kernel<<<gridGem, 128, 0, stream>>>(X, nullptr, nullptr, nullptr, Wp, O, 0);
  };
  auto gemm_recur = [&](const float* T, const float* P, float* Xo, const unsigned int* Wp, float* O) {
    cheb_gemm_kernel<<<gridGem, 128, 0, stream>>>(nullptr, T, P, Xo, Wp, O, 1);
  };
  // Full 5-term Chebyshev conv: X0 may be read-only (residual path); A/Bb/Cc are scratch.
  auto cheb_conv = [&](const float* X0, const unsigned int* Wp, float* O) {
    gemm_direct(X0, Wp + 0 * kWp, O);               // T0
    lmul(X0, Cc);                                   // Cc = T1 = L X0
    gemm_direct(Cc, Wp + 1 * kWp, O);
    lmul(Cc, Bb);                                   // Bb = L T1
    gemm_recur(Bb, X0, A, Wp + 2 * kWp, O);         // A  = T2 = 2 L T1 - T0
    lmul(A, Bb);                                    // Bb = L T2
    gemm_recur(Bb, Cc, Cc, Wp + 3 * kWp, O);        // Cc = T3 = 2 L T2 - T1
    lmul(Cc, Bb);                                   // Bb = L T3
    gemm_recur(Bb, A, A, Wp + 4 * kWp, O);          // A  = T4 = 2 L T3 - T2
  };

  // one-time weight pre-swizzle + time MLPs (tiny)
  pack_w_kernel<<<5, 128, 0, stream>>>(w_conv1, wp1);
  pack_w_kernel<<<5, 128, 0, stream>>>(w_conv2, wp2);
  pack_w_kernel<<<5, 128, 0, stream>>>(w_res,  wpr);
  time_mlp_kernel<<<kB, 128, 0, stream>>>(time_emb, w_mlp1, b_mlp1, w_mlp2, b_mlp2, t1, t2);

  // h = mish(bn1(x + t1)) -> A
  hipMemsetAsync(stats, 0, 2 * kC * sizeof(float), stream);
  bn_stats_kernel<<<gridBN, 128, 0, stream>>>(x, t1, stats);
  bn_apply_kernel<<<gridE4, 256, 0, stream>>>(x, t1, stats, gamma1, beta1, A);

  // conv1 -> O1
  bias_init_kernel<<<gridE4, 256, 0, stream>>>(O1, b_conv1, nullptr);
  cheb_conv(A, wp1, O1);

  // h = mish(bn2(O1 + t2)) -> A
  hipMemsetAsync(stats, 0, 2 * kC * sizeof(float), stream);
  bn_stats_kernel<<<gridBN, 128, 0, stream>>>(O1, t2, stats);
  bn_apply_kernel<<<gridE4, 256, 0, stream>>>(O1, t2, stats, gamma2, beta2, A);

  // out = b_conv2 + b_res + cheb_conv2(h) + cheb_conv_res(x)
  bias_init_kernel<<<gridE4, 256, 0, stream>>>(out, b_conv2, b_res);
  cheb_conv(A, wp2, out);
  cheb_conv(x, wpr, out);
}
